// WLN_Layer_80393197846861
// MI455X (gfx1250) — compile-verified, matrix operand order checked by reading, not run on hardware
//
#include <hip/hip_runtime.h>

typedef __attribute__((ext_vector_type(16))) __bf16 v16bf;
typedef __attribute__((ext_vector_type(8)))  float  v8f;

#define HD    128
#define NBMAX 10
#define ADIM  1024
#define EDIM  2048
#define NAT   65536   /* B*A */

static __device__ __forceinline__ v8f v8f_zero() {
  v8f z = {0.f, 0.f, 0.f, 0.f, 0.f, 0.f, 0.f, 0.f};
  return z;
}

static __device__ __forceinline__ v8f wmma_bf16(v16bf a, v16bf b, v8f c) {
  // D(16x16 f32) = A(16x32 bf16) x B(32x16 bf16) + C
  return __builtin_amdgcn_wmma_f32_16x16x32_bf16(false, a, false, b, (short)0, c,
                                                 false, false);
}

// Pre-swizzle a [K x 128] f32 weight (row-major, N=128) into bf16 WMMA
// B-fragment order in LDS: linear t = ((kk*8 + nn)*32 + lane)*16 + e, so a
// B-fragment load is one contiguous 32-byte LDS read per lane.
static __device__ __forceinline__ void load_weight_swz(const float* __restrict__ W,
                                                       __bf16* dst, int kSteps, int kMax,
                                                       int tid, int nthreads) {
  int total = kSteps * 8 * 32 * 16;
  for (int t = tid; t < total; t += nthreads) {
    int e    = t & 15;
    int lane = (t >> 4) & 31;
    int nn   = (t >> 9) & 7;
    int kk   = t >> 12;
    int k = kk * 32 + ((e >> 3) << 4) + (((lane >> 4) & 1) << 3) + (e & 7);
    int n = nn * 16 + (lane & 15);
    float v = (k < kMax) ? W[(size_t)k * HD + n] : 0.0f;
    dst[t] = (__bf16)v;
  }
}

static __device__ __forceinline__ v16bf lds_bfrag(const __bf16* buf, int kk, int nn,
                                                  int lane) {
  return *(const v16bf*)(buf + ((((kk << 3) + nn) * 32 + lane) << 4));
}

// Build an A-fragment (16x32 bf16, one K-step) from 16 row-major f32 rows.
static __device__ __forceinline__ v16bf frag_from_rows(const float* rows, int stride,
                                                       int kk, int lane) {
  int m = lane & 15, half = (lane >> 4) & 1;
  v16bf a;
  #pragma unroll
  for (int e = 0; e < 16; ++e) {
    int k = kk * 32 + ((e >> 3) << 4) + (half << 3) + (e & 7);
    a[e] = (__bf16)rows[m * stride + k];
  }
  return a;
}

static __device__ __forceinline__ v16bf frag_from_global(const float* __restrict__ row,
                                                         int kk, int lane) {
  int half = (lane >> 4) & 1;
  v16bf a;
  #pragma unroll
  for (int e = 0; e < 16; ++e) {
    int k = kk * 32 + ((e >> 3) << 4) + (half << 3) + (e & 7);
    a[e] = (__bf16)row[k];
  }
  return a;
}

// ---------------------------------------------------------------------------
// Kernel A: atom_features = input_atom @ W_atom   ([NAT,82] @ [82,128], K->96)
// ---------------------------------------------------------------------------
__global__ __launch_bounds__(256) void k_init_atom(const float* __restrict__ in_atom,
                                                   const float* __restrict__ W_atom,
                                                   float* __restrict__ feat) {
  __shared__ __align__(32) __bf16 Wswz[3 * 8 * 32 * 16];   // 24 KB
  __shared__ __align__(16) float  rows[8][16 * 96];        // 48 KB
  int tid = threadIdx.x, lane = tid & 31, wave = tid >> 5;

  load_weight_swz(W_atom, Wswz, 3, 82, tid, 256);

  int row0 = (blockIdx.x * 8 + wave) * 16;
  float* R = rows[wave];
  for (int i = lane; i < 16 * 96; i += 32) {
    int r = i / 96, k = i - r * 96;
    R[i] = (k < 82) ? in_atom[(size_t)(row0 + r) * 82 + k] : 0.f;
  }
  __syncthreads();

  v16bf afr[3];
  #pragma unroll
  for (int kk = 0; kk < 3; ++kk) afr[kk] = frag_from_rows(R, 96, kk, lane);

  int nl = lane & 15, half = (lane >> 4) & 1;
  #pragma unroll
  for (int nn = 0; nn < 8; ++nn) {
    v8f acc = v8f_zero();
    #pragma unroll
    for (int kk = 0; kk < 3; ++kk)
      acc = wmma_bf16(afr[kk], lds_bfrag(Wswz, kk, nn, lane), acc);
    int n = nn * 16 + nl;
    #pragma unroll
    for (int r = 0; r < 8; ++r) {
      int m = r + (half << 3);
      feat[(size_t)(row0 + m) * HD + n] = acc[r];
    }
  }
}

// ---------------------------------------------------------------------------
// Kernel B: per atom (one wave, NB padded to 16 rows):
//   P = gather(feat) @ W_nei_atom     (WMMA K=128)
//   Q = gather(feat) @ W_U2[:128]     (WMMA K=128, shared A-fragments)
//   hb = fbond @ W_nei_bond, qb = fbond @ W_U2[128:134]  (VALU, K=6)
//   nei_label = sum_m<nbs (Q + qb + b_U2);  f_nei = sum_m<nbs P*hb (last depth)
// ---------------------------------------------------------------------------
__global__ __launch_bounds__(256) void k_neighbor(
    const float* __restrict__ feat, const float* __restrict__ in_bond,
    const int* __restrict__ atom_graph, const int* __restrict__ bond_graph,
    const int* __restrict__ num_nbs,
    const float* __restrict__ W_nei_atom, const float* __restrict__ W_nei_bond,
    const float* __restrict__ W_U2, const float* __restrict__ b_U2,
    float* __restrict__ nei_label, float* __restrict__ f_nei, int want_fnei) {
  __shared__ __align__(32) __bf16 WnaS[4 * 8 * 32 * 16];   // 32 KB
  __shared__ __align__(32) __bf16 WqS[4 * 8 * 32 * 16];    // 32 KB
  __shared__ float Wnb[6 * HD];                            // 3 KB
  __shared__ float Wqb[6 * HD];                            // 3 KB
  __shared__ float bU2[HD];
  __shared__ __align__(16) float rows[8][16 * HD];         // 64 KB
  __shared__ float fb[8][16 * 6];                          // 3 KB
  int tid = threadIdx.x, lane = tid & 31, wave = tid >> 5;

  load_weight_swz(W_nei_atom, WnaS, 4, 128, tid, 256);
  load_weight_swz(W_U2, WqS, 4, 128, tid, 256);
  for (int t = tid; t < 6 * HD; t += 256) {
    Wnb[t] = W_nei_bond[t];
    Wqb[t] = W_U2[128 * HD + t];
  }
  for (int t = tid; t < HD; t += 256) bU2[t] = b_U2[t];

  int a = blockIdx.x * 8 + wave;   // atom in [0, NAT)
  int b = a >> 10;                 // batch index (A = 1024)
  const int* ag = atom_graph + (size_t)a * NBMAX;
  const int* bg = bond_graph + (size_t)a * NBMAX;

  float* R = rows[wave];
  for (int i = lane; i < 16 * HD; i += 32) {
    int m = i >> 7, k = i & 127;
    int idx = (m < NBMAX) ? ag[m] : 0;
    R[i] = feat[(size_t)(b * ADIM + idx) * HD + k];
  }
  float* FB = fb[wave];
  for (int i = lane; i < 16 * 6; i += 32) {
    int m = i / 6, t = i - m * 6;
    int e = (m < NBMAX) ? bg[m] : 0;
    FB[i] = in_bond[(size_t)(b * EDIM + e) * 6 + t];
  }
  __syncthreads();

  int nbs = num_nbs[a];
  v16bf afr[4];
  #pragma unroll
  for (int kk = 0; kk < 4; ++kk) afr[kk] = frag_from_rows(R, HD, kk, lane);

  int nl = lane & 15, half = (lane >> 4) & 1;
  #pragma unroll
  for (int nn = 0; nn < 8; ++nn) {
    v8f accP = v8f_zero();
    v8f accQ = v8f_zero();
    #pragma unroll
    for (int kk = 0; kk < 4; ++kk) {
      accP = wmma_bf16(afr[kk], lds_bfrag(WnaS, kk, nn, lane), accP);
      accQ = wmma_bf16(afr[kk], lds_bfrag(WqS, kk, nn, lane), accQ);
    }
    int n = nn * 16 + nl;
    float wnb[6], wqb[6];
    #pragma unroll
    for (int t = 0; t < 6; ++t) { wnb[t] = Wnb[t * HD + n]; wqb[t] = Wqb[t * HD + n]; }
    float bias = bU2[n];
    float sP = 0.f, sQ = 0.f;
    #pragma unroll
    for (int r = 0; r < 8; ++r) {
      int m = r + (half << 3);
      if (m < nbs) {
        float hb = 0.f, qb = 0.f;
        #pragma unroll
        for (int t = 0; t < 6; ++t) {
          float f = FB[m * 6 + t];
          hb += f * wnb[t];
          qb += f * wqb[t];
        }
        sP += accP[r] * hb;
        sQ += accQ[r] + qb + bias;
      }
    }
    sP += __shfl_xor(sP, 16, 32);
    sQ += __shfl_xor(sQ, 16, 32);
    if (half == 0) {
      nei_label[(size_t)a * HD + n] = sQ;
      if (want_fnei) f_nei[(size_t)a * HD + n] = sP;
    }
  }
}

// ---------------------------------------------------------------------------
// Kernel C: feat_out = feat @ W_U1a + nei_label @ W_U1b + b_U1
//           (last depth) kernel = f_nei * (feat @ W_self) * node_mask
// ---------------------------------------------------------------------------
__global__ __launch_bounds__(256) void k_update(
    const float* __restrict__ feat, const float* __restrict__ nlab,
    const float* __restrict__ f_nei_in, const float* __restrict__ node_mask,
    const float* __restrict__ W_self, const float* __restrict__ W_U1,
    const float* __restrict__ b_U1, float* __restrict__ feat_out,
    float* __restrict__ kernel_out, int want_kernel) {
  __shared__ __align__(32) __bf16 WsS[4 * 8 * 32 * 16];    // 32 KB
  __shared__ __align__(32) __bf16 Wu1aS[4 * 8 * 32 * 16];  // 32 KB
  __shared__ __align__(32) __bf16 Wu1bS[4 * 8 * 32 * 16];  // 32 KB
  __shared__ float bU1[HD];
  int tid = threadIdx.x, lane = tid & 31, wave = tid >> 5;

  if (want_kernel) load_weight_swz(W_self, WsS, 4, 128, tid, 256);
  load_weight_swz(W_U1, Wu1aS, 4, 128, tid, 256);
  load_weight_swz(W_U1 + 128 * HD, Wu1bS, 4, 128, tid, 256);
  for (int t = tid; t < HD; t += 256) bU1[t] = b_U1[t];
  __syncthreads();

  int row0 = (blockIdx.x * 8 + wave) * 16;
  int nl = lane & 15, half = (lane >> 4) & 1;
  const float* frow = feat + (size_t)(row0 + nl) * HD;
  const float* nrow = nlab + (size_t)(row0 + nl) * HD;

  v16bf af[4], nf[4];
  #pragma unroll
  for (int kk = 0; kk < 4; ++kk) {
    af[kk] = frag_from_global(frow, kk, lane);
    nf[kk] = frag_from_global(nrow, kk, lane);
  }

  #pragma unroll
  for (int nn = 0; nn < 8; ++nn) {
    v8f accU = v8f_zero();
    #pragma unroll
    for (int kk = 0; kk < 4; ++kk) {
      accU = wmma_bf16(af[kk], lds_bfrag(Wu1aS, kk, nn, lane), accU);
      accU = wmma_bf16(nf[kk], lds_bfrag(Wu1bS, kk, nn, lane), accU);
    }
    int n = nn * 16 + nl;
    float bias = bU1[n];
    v8f accS = v8f_zero();
    if (want_kernel) {
      #pragma unroll
      for (int kk = 0; kk < 4; ++kk)
        accS = wmma_bf16(af[kk], lds_bfrag(WsS, kk, nn, lane), accS);
    }
    #pragma unroll
    for (int r = 0; r < 8; ++r) {
      int m = r + (half << 3);
      size_t o = (size_t)(row0 + m) * HD + n;
      feat_out[o] = accU[r] + bias;
      if (want_kernel) kernel_out[o] = f_nei_in[o] * accS[r] * node_mask[row0 + m];
    }
  }
}

// ---------------------------------------------------------------------------
extern "C" void kernel_launch(void* const* d_in, const int* in_sizes, int n_in,
                              void* d_out, int out_size, void* d_ws, size_t ws_size,
                              hipStream_t stream) {
  (void)in_sizes; (void)n_in; (void)out_size; (void)ws_size;
  const float* input_atom = (const float*)d_in[0];
  const float* input_bond = (const float*)d_in[1];
  const int*   atom_graph = (const int*)d_in[2];
  const int*   bond_graph = (const int*)d_in[3];
  const int*   num_nbs    = (const int*)d_in[4];
  const float* node_mask  = (const float*)d_in[5];
  const float* W_atom     = (const float*)d_in[6];
  const float* W_nei_atom = (const float*)d_in[7];
  const float* W_nei_bond = (const float*)d_in[8];
  const float* W_self     = (const float*)d_in[9];
  const float* W_U2       = (const float*)d_in[10];
  const float* b_U2       = (const float*)d_in[11];
  const float* W_U1       = (const float*)d_in[12];
  const float* b_U1       = (const float*)d_in[13];

  const size_t FEAT_ELEMS = (size_t)NAT * HD;      // 8M floats = 32 MB
  float* out_kernel = (float*)d_out;               // first half of d_out
  float* out_feat   = (float*)d_out + FEAT_ELEMS;  // second half (also ping buf F1)
  float* F0   = (float*)d_ws;                      // ws[0,32MB)
  float* NLAB = F0 + FEAT_ELEMS;                   // ws[32MB,64MB)
  float* F1   = out_feat;

  dim3 blk(256);
  dim3 gTiles(NAT / 16 / 8);   // 512 blocks (16 rows per wave, 8 waves)
  dim3 gAtoms(NAT / 8);        // 8192 blocks (1 atom per wave)

  k_init_atom<<<gTiles, blk, 0, stream>>>(input_atom, W_atom, F0);

  // depth 0: F0 -> F1
  k_neighbor<<<gAtoms, blk, 0, stream>>>(F0, input_bond, atom_graph, bond_graph,
                                         num_nbs, W_nei_atom, W_nei_bond, W_U2, b_U2,
                                         NLAB, out_kernel, 0);
  k_update<<<gTiles, blk, 0, stream>>>(F0, NLAB, out_kernel, node_mask, W_self, W_U1,
                                       b_U1, F1, out_kernel, 0);
  // depth 1: F1 -> F0
  k_neighbor<<<gAtoms, blk, 0, stream>>>(F1, input_bond, atom_graph, bond_graph,
                                         num_nbs, W_nei_atom, W_nei_bond, W_U2, b_U2,
                                         NLAB, out_kernel, 0);
  k_update<<<gTiles, blk, 0, stream>>>(F1, NLAB, out_kernel, node_mask, W_self, W_U1,
                                       b_U1, F0, out_kernel, 0);
  // depth 2 (last): F0 -> out_feat; f_nei + kernel into out_kernel
  k_neighbor<<<gAtoms, blk, 0, stream>>>(F0, input_bond, atom_graph, bond_graph,
                                         num_nbs, W_nei_atom, W_nei_bond, W_U2, b_U2,
                                         NLAB, out_kernel, 1);
  k_update<<<gTiles, blk, 0, stream>>>(F0, NLAB, out_kernel, node_mask, W_self, W_U1,
                                       b_U1, out_feat, out_kernel, 1);
}
